// SelectorModule_27650999451839
// MI455X (gfx1250) — matrix operand order repaired and checked by reading, not compiled, not run on hardware
//
#include <hip/hip_runtime.h>

// ---------------------------------------------------------------------------
// MI455X (gfx1250) implementation of the SelectorModule multi-modal attention.
// All GEMMs use v_wmma_f32_16x16x32_bf16 (wave32). A-tiles are staged into
// LDS with gfx1250 async global->LDS copies (ASYNCcnt), double-buffered.
// ---------------------------------------------------------------------------

#define Bd 8
#define Td 512
#define Dd 512
#define Hd 8
#define DKd 128
#define DVd 128
#define Md 2
#define Sd 2
#define Gd 4
#define Ld 512
#define MSd 4          // M*S
#define LTOT 2048      // MS*L

#define AS1 __attribute__((address_space(1)))
#define AS3 __attribute__((address_space(3)))

#if defined(__has_builtin)
#if __has_builtin(__builtin_amdgcn_global_load_async_to_lds_b128) && \
    __has_builtin(__builtin_amdgcn_s_wait_asynccnt)
#define HAVE_ASYNC_LDS 1
#endif
#endif
#ifndef HAVE_ASYNC_LDS
#define HAVE_ASYNC_LDS 0
#endif

typedef __attribute__((ext_vector_type(16))) __bf16 v16bf;
typedef __attribute__((ext_vector_type(8)))  float  v8f;
typedef __attribute__((ext_vector_type(4)))  int    v4i_;

union FragU { unsigned int w[8]; v16bf v; };

__device__ inline unsigned short f32_to_bf16(float f) {
    unsigned int u = __float_as_uint(f);
    u += 0x7FFFu + ((u >> 16) & 1u);          // round to nearest even
    return (unsigned short)(u >> 16);
}
__device__ inline float bf16_to_f32(unsigned short h) {
    return __uint_as_float(((unsigned int)h) << 16);
}
__device__ inline unsigned int pack_bf16x2(float lo, float hi) {
    return (unsigned int)f32_to_bf16(lo) | ((unsigned int)f32_to_bf16(hi) << 16);
}
__device__ inline v8f vzero8() {
    v8f r;
    #pragma unroll
    for (int i = 0; i < 8; ++i) r[i] = 0.0f;
    return r;
}

// 16-byte async (or fallback sync) copy: global -> LDS
__device__ inline void copy16_g2lds(const void* g, void* l) {
#if HAVE_ASYNC_LDS
    __builtin_amdgcn_global_load_async_to_lds_b128(
        (AS1 v4i_*)(v4i_*)(void*)g,      // global source
        (AS3 v4i_*)(v4i_*)l,             // LDS destination
        0, 0);
#else
    *(uint4*)l = *(const uint4*)g;
#endif
}
__device__ inline void async_wait0() {
#if HAVE_ASYNC_LDS
    __builtin_amdgcn_s_wait_asynccnt(0);
#endif
}
__device__ inline void async_wait1() {
#if HAVE_ASYNC_LDS
    __builtin_amdgcn_s_wait_asynccnt(1);
#endif
}
__device__ inline void async_wait2() {
#if HAVE_ASYNC_LDS
    __builtin_amdgcn_s_wait_asynccnt(2);
#endif
}

// ---- fragment loaders -----------------------------------------------------
// A fragment 16x32 bf16 from LDS tile (row stride in bytes, 16B aligned).
// ISA layout: VGPR j holds K = 16*(j>>2) + 2*(j&3) + 8*hi (+next), hi=lane>>4.
__device__ inline FragU lds_a_bf16(const char* base, int strideBytes, int m0rel) {
    int lane = threadIdx.x & 31;
    int row  = m0rel + (lane & 15);
    int hi   = lane >> 4;
    const char* rp = base + row * strideBytes;
    FragU f;
    #pragma unroll
    for (int j = 0; j < 8; ++j) {
        int k = ((j >> 2) << 4) + ((j & 3) << 1) + (hi << 3);
        f.w[j] = *(const unsigned int*)(rp + k * 2);
    }
    return f;
}
// A fragment from LDS tile of f32 (converted to bf16 on read).
__device__ inline FragU lds_a_f32cvt(const char* base, int strideBytes, int m0rel) {
    int lane = threadIdx.x & 31;
    int row  = m0rel + (lane & 15);
    int hi   = lane >> 4;
    const float* rp = (const float*)(base + row * strideBytes);
    FragU f;
    #pragma unroll
    for (int j = 0; j < 8; ++j) {
        int k = ((j >> 2) << 4) + ((j & 3) << 1) + (hi << 3);
        f.w[j] = pack_bf16x2(rp[k], rp[k + 1]);
    }
    return f;
}
// B fragment, 32x16 bf16, stored n-major: Bt[n][k], from global.
// ISA layout: lane n = l&15, VGPR j holds K = 16*(j>>2) + 2*(j&3) + 8*hi (+next).
__device__ inline FragU load_b_nmajor(const unsigned short* Bt, int ldk, int n0, int k0) {
    int lane = threadIdx.x & 31;
    int n    = n0 + (lane & 15);
    int hi   = lane >> 4;
    const unsigned int* p = (const unsigned int*)(Bt + (long)n * ldk + k0);
    FragU f;
    #pragma unroll
    for (int j = 0; j < 8; ++j) f.w[j] = p[((j >> 2) << 3) + (j & 3) + (hi << 2)];
    return f;
}

__device__ inline v8f wmma_bf16(FragU a, FragU b, v8f c) {
    return __builtin_amdgcn_wmma_f32_16x16x32_bf16(false, a.v, false, b.v,
                                                   (short)0, c, false, false);
}

// reductions within each 16-lane half (xor masks < 16 keep halves separate)
__device__ inline float hmax16(float v) {
    v = fmaxf(v, __shfl_xor(v, 1, 32));
    v = fmaxf(v, __shfl_xor(v, 2, 32));
    v = fmaxf(v, __shfl_xor(v, 4, 32));
    v = fmaxf(v, __shfl_xor(v, 8, 32));
    return v;
}
__device__ inline float hsum16(float v) {
    v += __shfl_xor(v, 1, 32);
    v += __shfl_xor(v, 2, 32);
    v += __shfl_xor(v, 4, 32);
    v += __shfl_xor(v, 8, 32);
    return v;
}

#define A_STRIDE   80     // bf16 tile: 64 rows x 32 k (64B data), 16B aligned, bank-clean
#define A_STRIDE_F 144    // f32 tile:  64 rows x 32 k (128B data), 16B aligned, bank-clean

// ---------------------------------------------------------------------------
__global__ void k_transpose_bf16(const float* __restrict__ W,
                                 unsigned short* __restrict__ Wt, int K, int N) {
    long base = (long)blockIdx.z * K * N;
    for (long i = blockIdx.x * (long)blockDim.x + threadIdx.x; i < (long)K * N;
         i += (long)gridDim.x * blockDim.x) {
        int k = (int)(i / N), n = (int)(i % N);
        Wt[base + (long)n * K + k] = f32_to_bf16(W[base + i]);
    }
}

__global__ void k_to_bf16(const float* __restrict__ s, unsigned short* __restrict__ d, long n) {
    for (long i = blockIdx.x * (long)blockDim.x + threadIdx.x; i < n;
         i += (long)gridDim.x * blockDim.x)
        d[i] = f32_to_bf16(s[i]);
}

// ---------------------------------------------------------------------------
__global__ __launch_bounds__(256) void k_layernorm(const float* __restrict__ x,
                                                   const float* __restrict__ gamma,
                                                   const float* __restrict__ beta,
                                                   unsigned short* __restrict__ out) {
    int row = blockIdx.x;
    const float* xr = x + (long)row * Dd;
    int tid = threadIdx.x, lane = tid & 31, wave = tid >> 5;
    float v0 = xr[tid], v1 = xr[tid + 256];
    float s = v0 + v1, s2 = v0 * v0 + v1 * v1;
    #pragma unroll
    for (int m = 1; m < 32; m <<= 1) { s += __shfl_xor(s, m, 32); s2 += __shfl_xor(s2, m, 32); }
    __shared__ float rs[8], rs2[8];
    if (lane == 0) { rs[wave] = s; rs2[wave] = s2; }
    __syncthreads();
    float ts = 0.f, ts2 = 0.f;
    #pragma unroll
    for (int w = 0; w < 8; ++w) { ts += rs[w]; ts2 += rs2[w]; }
    float mu  = ts * (1.0f / Dd);
    float var = ts2 * (1.0f / Dd) - mu * mu;
    float inv = rsqrtf(var + 1e-6f);
    out[(long)row * Dd + tid]       = f32_to_bf16((v0 - mu) * inv * gamma[tid]       + beta[tid]);
    out[(long)row * Dd + tid + 256] = f32_to_bf16((v1 - mu) * inv * gamma[tid + 256] + beta[tid + 256]);
}

// ---------------------------------------------------------------------------
// Generic projection GEMM: C[rows x 1024] = A[rows x 512](bf16,rowmajor)
//                                         x Bt[1024 x 512](bf16,n-major)
// grid = (8, rows/64, batch); block = 256 (8 waves). Block tile 64x128.
// A tile staged through LDS with async copies, double buffered (16 k-steps).
// ---------------------------------------------------------------------------
__global__ __launch_bounds__(256) void k_proj(const unsigned short* __restrict__ A, long strideA,
                                              const unsigned short* __restrict__ Bt, long strideB,
                                              unsigned short* __restrict__ out, int mode) {
    int z = blockIdx.z;
    const unsigned short* Az  = A  + (long)z * strideA;
    const unsigned short* Btz = Bt + (long)z * strideB;
    int nblk = blockIdx.x, mblk = blockIdx.y;
    int wave = threadIdx.x >> 5, lane = threadIdx.x & 31, hi = lane >> 4;
    int n0 = nblk * 128 + wave * 16;

    __shared__ char sA[2][64 * A_STRIDE];
    int cr = threadIdx.x >> 2, cc = threadIdx.x & 3;   // copy row / 16B chunk

    v8f acc[4];
    #pragma unroll
    for (int i = 0; i < 4; ++i) acc[i] = vzero8();

    // prologue: stage k-step 0
    copy16_g2lds(Az + (long)(mblk * 64 + cr) * 512 + cc * 8, &sA[0][cr * A_STRIDE + cc * 16]);

    for (int ks = 0; ks < 15; ++ks) {
        copy16_g2lds(Az + (long)(mblk * 64 + cr) * 512 + (ks + 1) * 32 + cc * 8,
                     &sA[(ks + 1) & 1][cr * A_STRIDE + cc * 16]);
        async_wait1();
        __syncthreads();
        FragU bfr = load_b_nmajor(Btz, 512, n0, ks * 32);
        const char* Abuf = sA[ks & 1];
        #pragma unroll
        for (int mt = 0; mt < 4; ++mt)
            acc[mt] = wmma_bf16(lds_a_bf16(Abuf, A_STRIDE, mt * 16), bfr, acc[mt]);
        __syncthreads();
    }
    async_wait0();
    __syncthreads();
    {
        FragU bfr = load_b_nmajor(Btz, 512, n0, 15 * 32);
        const char* Abuf = sA[1];
        #pragma unroll
        for (int mt = 0; mt < 4; ++mt)
            acc[mt] = wmma_bf16(lds_a_bf16(Abuf, A_STRIDE, mt * 16), bfr, acc[mt]);
    }

    int nn = n0 + (lane & 15);
    int hh = nn >> 7, dk = nn & 127;
    #pragma unroll
    for (int mt = 0; mt < 4; ++mt) {
        #pragma unroll
        for (int r = 0; r < 8; ++r) {
            int row = mblk * 64 + mt * 16 + r + hi * 8;
            float v = acc[mt][r];
            if (mode == 0) {                      // row = b*T + t
                int b = row >> 9, t = row & 511;
                out[((long)(b * Hd + hh) * Td + t) * DKd + dk] =
                    f32_to_bf16(v * 0.08838834764831845f);   // 1/sqrt(128)
            } else {                              // row = s*B*L + b*L + l
                int s = row >> 12, b = (row >> 9) & 7, l = row & 511;
                int ms = z * Sd + s;
                if (mode == 1)
                    out[(((long)((ms * Bd + b) * Hd + hh)) * Ld + l) * DKd + dk] = f32_to_bf16(v);
                else
                    out[(((long)((ms * Bd + b) * Hd + hh)) * DVd + dk) * Ld + l] = f32_to_bf16(v);
            }
        }
    }
}

// ---------------------------------------------------------------------------
// Guider keys (tiny, scalar f32): gk[b][h][g][dk]
// ---------------------------------------------------------------------------
__global__ __launch_bounds__(256) void k_gk(const float* __restrict__ guider,
                                            const float* __restrict__ Wkg,
                                            float* __restrict__ gk) {
    int bg = blockIdx.x;
    int b = bg >> 2, g = bg & 3, m = g >> 1;
    __shared__ float gx[Dd];
    gx[threadIdx.x]       = guider[((long)b * Gd + g) * Dd + threadIdx.x];
    gx[threadIdx.x + 256] = guider[((long)b * Gd + g) * Dd + threadIdx.x + 256];
    __syncthreads();
    const float* W = Wkg + (long)m * Dd * (Hd * DKd);
    for (int n = threadIdx.x; n < Hd * DKd; n += 256) {
        float acc = 0.f;
        for (int d = 0; d < Dd; ++d) acc += gx[d] * W[(long)d * (Hd * DKd) + n];
        int h = n >> 7, dk = n & 127;
        gk[(((long)b * Hd + h) * Gd + g) * DKd + dk] = acc;
    }
}

// ---------------------------------------------------------------------------
// Guider attention softmax (over G=4): one thread per (b,h,t)
// ---------------------------------------------------------------------------
__global__ __launch_bounds__(256) void k_gatt(const unsigned short* __restrict__ q,
                                              const float* __restrict__ gk,
                                              const int* __restrict__ gmask,
                                              float* __restrict__ gatt) {
    int idx = blockIdx.x * 256 + threadIdx.x;       // b*H*T
    if (idx >= Bd * Hd * Td) return;
    int t = idx & 511, h = (idx >> 9) & 7, b = idx >> 12;
    const unsigned short* qr = q + ((long)(b * Hd + h) * Td + t) * DKd;
    const float* gr = gk + (long)(b * Hd + h) * Gd * DKd;
    float s[4] = {0.f, 0.f, 0.f, 0.f};
    for (int d = 0; d < DKd; ++d) {
        float qv = bf16_to_f32(qr[d]);
        s[0] += qv * gr[d];
        s[1] += qv * gr[DKd + d];
        s[2] += qv * gr[2 * DKd + d];
        s[3] += qv * gr[3 * DKd + d];
    }
    const int* mk = gmask + ((long)b * Td + t) * Gd;
    float mx = -3.0e38f;
    #pragma unroll
    for (int g = 0; g < 4; ++g) { if (mk[g] == 0) s[g] = -1e9f; mx = fmaxf(mx, s[g]); }
    float sum = 0.f;
    #pragma unroll
    for (int g = 0; g < 4; ++g) { s[g] = __expf(s[g] - mx); sum += s[g]; }
    float inv = 1.0f / sum;
    #pragma unroll
    for (int g = 0; g < 4; ++g)
        gatt[(((long)(b * Hd + h) * Td + t)) * Gd + g] = s[g] * inv;
}

// ---------------------------------------------------------------------------
// Fused low-level scores + masked softmax + guider weighting.
// grid = (T/64, H, MS*B); block 256 (8 waves). Q tile async-staged via LDS.
// ---------------------------------------------------------------------------
__global__ __launch_bounds__(256) void k_scores(const unsigned short* __restrict__ q,
                                                const unsigned short* __restrict__ lk,
                                                const float* __restrict__ gatt,
                                                const int* __restrict__ lmask,
                                                float* __restrict__ fatt) {
    int tblk = blockIdx.x, h = blockIdx.y;
    int ms = blockIdx.z >> 3, b = blockIdx.z & 7;
    int wave = threadIdx.x >> 5, lane = threadIdx.x & 31, hi = lane >> 4;
    int t0 = tblk * 64;

    const unsigned short* qb = q  + ((long)(b * Hd + h) * Td) * DKd;
    const unsigned short* kb = lk + ((long)((ms * Bd + b) * Hd + h) * Ld) * DKd;

    __shared__ char sA[2][64 * A_STRIDE];
    __shared__ float red[64][8];
    int cr = threadIdx.x >> 2, cc = threadIdx.x & 3;

    v8f acc[4][4];
    #pragma unroll
    for (int mt = 0; mt < 4; ++mt)
        #pragma unroll
        for (int nt = 0; nt < 4; ++nt) acc[mt][nt] = vzero8();

    copy16_g2lds(qb + (long)(t0 + cr) * DKd + cc * 8, &sA[0][cr * A_STRIDE + cc * 16]);

    #pragma unroll
    for (int ks = 0; ks < 4; ++ks) {
        if (ks < 3)
            copy16_g2lds(qb + (long)(t0 + cr) * DKd + (ks + 1) * 32 + cc * 8,
                         &sA[(ks + 1) & 1][cr * A_STRIDE + cc * 16]);
        if (ks < 3) async_wait1(); else async_wait0();
        __syncthreads();
        const char* Abuf = sA[ks & 1];
        FragU a[4];
        #pragma unroll
        for (int mt = 0; mt < 4; ++mt) a[mt] = lds_a_bf16(Abuf, A_STRIDE, mt * 16);
        #pragma unroll
        for (int nt = 0; nt < 4; ++nt) {
            FragU bfr = load_b_nmajor(kb, DKd, wave * 64 + nt * 16, ks * 32);
            #pragma unroll
            for (int mt = 0; mt < 4; ++mt)
                acc[mt][nt] = wmma_bf16(a[mt], bfr, acc[mt][nt]);
        }
        __syncthreads();
    }

    // apply low_mask
    const int* mrow = lmask + ((long)(ms * Bd + b) * Td) * Ld;
    #pragma unroll
    for (int mt = 0; mt < 4; ++mt)
        #pragma unroll
        for (int nt = 0; nt < 4; ++nt) {
            int col = wave * 64 + nt * 16 + (lane & 15);
            #pragma unroll
            for (int r = 0; r < 8; ++r) {
                int t = t0 + mt * 16 + r + hi * 8;
                if (mrow[(long)t * Ld + col] == 0) acc[mt][nt][r] = -1e9f;
            }
        }

    float rstat[4][8];

    // row max: wave-local over 4 n-tiles + 16-lane butterfly; cross-wave via LDS
    #pragma unroll
    for (int mt = 0; mt < 4; ++mt)
        #pragma unroll
        for (int r = 0; r < 8; ++r) {
            float v = fmaxf(fmaxf(acc[mt][0][r], acc[mt][1][r]),
                            fmaxf(acc[mt][2][r], acc[mt][3][r]));
            v = hmax16(v);
            if ((lane & 15) == 0) red[mt * 16 + r + hi * 8][wave] = v;
        }
    __syncthreads();
    #pragma unroll
    for (int mt = 0; mt < 4; ++mt)
        #pragma unroll
        for (int r = 0; r < 8; ++r) {
            int rr = mt * 16 + r + hi * 8;
            float v = red[rr][0];
            #pragma unroll
            for (int w = 1; w < 8; ++w) v = fmaxf(v, red[rr][w]);
            rstat[mt][r] = v;
        }
    __syncthreads();

    // exp + row sum
    #pragma unroll
    for (int mt = 0; mt < 4; ++mt)
        #pragma unroll
        for (int r = 0; r < 8; ++r) {
            float s = 0.f;
            #pragma unroll
            for (int nt = 0; nt < 4; ++nt) {
                float e = __expf(acc[mt][nt][r] - rstat[mt][r]);
                acc[mt][nt][r] = e;
                s += e;
            }
            s = hsum16(s);
            if ((lane & 15) == 0) red[mt * 16 + r + hi * 8][wave] = s;
        }
    __syncthreads();

    // combine sums, fold in guider attention, write final_att
    #pragma unroll
    for (int mt = 0; mt < 4; ++mt)
        #pragma unroll
        for (int r = 0; r < 8; ++r) {
            int rr = mt * 16 + r + hi * 8;
            float s = 0.f;
            #pragma unroll
            for (int w = 0; w < 8; ++w) s += red[rr][w];
            int t = t0 + rr;
            float g = gatt[(((long)(b * Hd + h) * Td + t)) * Gd + ms];
            rstat[mt][r] = g / s;
        }
    #pragma unroll
    for (int mt = 0; mt < 4; ++mt)
        #pragma unroll
        for (int nt = 0; nt < 4; ++nt) {
            int col = wave * 64 + nt * 16 + (lane & 15);
            #pragma unroll
            for (int r = 0; r < 8; ++r) {
                int t = t0 + mt * 16 + r + hi * 8;
                fatt[((long)(b * Hd + h) * Td + t) * LTOT + ms * Ld + col] =
                    acc[mt][nt][r] * rstat[mt][r];
            }
        }
}

// ---------------------------------------------------------------------------
// AV GEMM: attn_out[b][t][h*DV+dv] (bf16) = final_att[b][h][T][2048] x V
// final_att (f32) async-staged through LDS, converted to bf16 on LDS read.
// grid = (T/64, H, B); block 256; wave w owns dv slice [w*16, w*16+16).
// ---------------------------------------------------------------------------
__global__ __launch_bounds__(256) void k_av(const float* __restrict__ fatt,
                                            const unsigned short* __restrict__ lvt,
                                            unsigned short* __restrict__ attn) {
    int tblk = blockIdx.x, h = blockIdx.y, b = blockIdx.z;
    int wave = threadIdx.x >> 5, lane = threadIdx.x & 31, hi = lane >> 4;
    int t0 = tblk * 64;
    const float* fb = fatt + ((long)(b * Hd + h) * Td) * LTOT;

    __shared__ char sA[2][64 * A_STRIDE_F];

    v8f acc[4];
    #pragma unroll
    for (int i = 0; i < 4; ++i) acc[i] = vzero8();

    // prologue: stage k-step 0 (each thread copies 2 x 16B)
    #pragma unroll
    for (int it = 0; it < 2; ++it) {
        int idx = threadIdx.x + it * 256, r = idx >> 3, c = idx & 7;
        copy16_g2lds(fb + (long)(t0 + r) * LTOT + c * 4, &sA[0][r * A_STRIDE_F + c * 16]);
    }

    for (int ks = 0; ks < 63; ++ks) {
        #pragma unroll
        for (int it = 0; it < 2; ++it) {
            int idx = threadIdx.x + it * 256, r = idx >> 3, c = idx & 7;
            copy16_g2lds(fb + (long)(t0 + r) * LTOT + (ks + 1) * 32 + c * 4,
                         &sA[(ks + 1) & 1][r * A_STRIDE_F + c * 16]);
        }
        async_wait2();
        __syncthreads();
        int kk = ks * 32;
        int msc = kk >> 9, l0 = kk & 511;
        const unsigned short* vb = lvt + ((long)((msc * Bd + b) * Hd + h)) * DVd * Ld;
        FragU bfr = load_b_nmajor(vb, Ld, wave * 16, l0);
        const char* Abuf = sA[ks & 1];
        #pragma unroll
        for (int mt = 0; mt < 4; ++mt)
            acc[mt] = wmma_bf16(lds_a_f32cvt(Abuf, A_STRIDE_F, mt * 16), bfr, acc[mt]);
        __syncthreads();
    }
    async_wait0();
    __syncthreads();
    {
        const unsigned short* vb = lvt + ((long)((3 * Bd + b) * Hd + h)) * DVd * Ld;
        FragU bfr = load_b_nmajor(vb, Ld, wave * 16, 63 * 32 - 3 * 512);
        const char* Abuf = sA[1];
        #pragma unroll
        for (int mt = 0; mt < 4; ++mt)
            acc[mt] = wmma_bf16(lds_a_f32cvt(Abuf, A_STRIDE_F, mt * 16), bfr, acc[mt]);
    }

    int dv = wave * 16 + (lane & 15);
    #pragma unroll
    for (int mt = 0; mt < 4; ++mt)
        #pragma unroll
        for (int r = 0; r < 8; ++r) {
            int t = t0 + mt * 16 + r + hi * 8;
            attn[((long)b * Td + t) * (Hd * DVd) + h * DVd + dv] = f32_to_bf16(acc[mt][r]);
        }
}

// ---------------------------------------------------------------------------
// Final projection + residual: out[row][n] = attn[row][:]1024 x Wfc + trg
// grid = (512/128, 4096/64); block 256. A tile async-staged via LDS.
// ---------------------------------------------------------------------------
__global__ __launch_bounds__(256) void k_wfc(const unsigned short* __restrict__ attn,
                                             const unsigned short* __restrict__ wfct,
                                             const float* __restrict__ resid,
                                             float* __restrict__ out) {
    int nblk = blockIdx.x, mblk = blockIdx.y;
    int wave = threadIdx.x >> 5, lane = threadIdx.x & 31, hi = lane >> 4;
    int n0 = nblk * 128 + wave * 16;

    __shared__ char sA[2][64 * A_STRIDE];
    int cr = threadIdx.x >> 2, cc = threadIdx.x & 3;

    v8f acc[4];
    #pragma unroll
    for (int i = 0; i < 4; ++i) acc[i] = vzero8();

    copy16_g2lds(attn + (long)(mblk * 64 + cr) * 1024 + cc * 8, &sA[0][cr * A_STRIDE + cc * 16]);

    for (int ks = 0; ks < 31; ++ks) {
        copy16_g2lds(attn + (long)(mblk * 64 + cr) * 1024 + (ks + 1) * 32 + cc * 8,
                     &sA[(ks + 1) & 1][cr * A_STRIDE + cc * 16]);
        async_wait1();
        __syncthreads();
        FragU bfr = load_b_nmajor(wfct, Hd * DVd, n0, ks * 32);
        const char* Abuf = sA[ks & 1];
        #pragma unroll
        for (int mt = 0; mt < 4; ++mt)
            acc[mt] = wmma_bf16(lds_a_bf16(Abuf, A_STRIDE, mt * 16), bfr, acc[mt]);
        __syncthreads();
    }
    async_wait0();
    __syncthreads();
    {
        FragU bfr = load_b_nmajor(wfct, Hd * DVd, n0, 31 * 32);
        const char* Abuf = sA[1];
        #pragma unroll
        for (int mt = 0; mt < 4; ++mt)
            acc[mt] = wmma_bf16(lds_a_bf16(Abuf, A_STRIDE, mt * 16), bfr, acc[mt]);
    }

    int nn = n0 + (lane & 15);
    #pragma unroll
    for (int mt = 0; mt < 4; ++mt)
        #pragma unroll
        for (int r = 0; r < 8; ++r) {
            int row = mblk * 64 + mt * 16 + r + hi * 8;
            out[(long)row * Dd + nn] = acc[mt][r] + resid[(long)row * Dd + nn];
        }
}

// ---------------------------------------------------------------------------
extern "C" void kernel_launch(void* const* d_in, const int* in_sizes, int n_in,
                              void* d_out, int out_size, void* d_ws, size_t ws_size,
                              hipStream_t stream) {
    const float* trg    = (const float*)d_in[0];
    const float* guider = (const float*)d_in[1];
    const float* low    = (const float*)d_in[2];
    const float* Wq     = (const float*)d_in[3];
    const float* Wkg    = (const float*)d_in[4];
    const float* Wkl    = (const float*)d_in[5];
    const float* Wvl    = (const float*)d_in[6];
    const float* Wfc    = (const float*)d_in[7];
    const float* gamma  = (const float*)d_in[8];
    const float* beta   = (const float*)d_in[9];
    const int*   gmask  = (const int*)d_in[10];
    const int*   lmask  = (const int*)d_in[11];

    float* out  = (float*)d_out;
    float* fatt = out + (size_t)Bd * Td * Dd;   // final_att part of the tuple

    char*  ws  = (char*)d_ws;
    size_t off = 0;
    auto alloc = [&](size_t bytes) -> char* {
        char* p = ws + off;
        off = (off + bytes + 255) & ~(size_t)255;
        return p;
    };

    unsigned short* xln  = (unsigned short*)alloc((size_t)Bd * Td * Dd * 2);
    unsigned short* qb   = (unsigned short*)alloc((size_t)Bd * Hd * Td * DKd * 2);
    unsigned short* lowb = (unsigned short*)alloc((size_t)MSd * Bd * Ld * Dd * 2);
    unsigned short* wqt  = (unsigned short*)alloc((size_t)1024 * 512 * 2);
    unsigned short* wklt = (unsigned short*)alloc((size_t)2 * 1024 * 512 * 2);
    unsigned short* wvlt = (unsigned short*)alloc((size_t)2 * 1024 * 512 * 2);
    unsigned short* wfct = (unsigned short*)alloc((size_t)512 * 1024 * 2);
    float*          gk   = (float*)alloc((size_t)Bd * Hd * Gd * DKd * 4);
    float*          gatt = (float*)alloc((size_t)Bd * Hd * Td * Gd * 4);
    unsigned short* lkb  = (unsigned short*)alloc((size_t)MSd * Bd * Hd * Ld * DKd * 2);
    unsigned short* lvt  = (unsigned short*)alloc((size_t)MSd * Bd * Hd * DVd * Ld * 2);
    unsigned short* attn = (unsigned short*)alloc((size_t)Bd * Td * Hd * DVd * 2);

    k_transpose_bf16<<<dim3(2048, 1, 1), 256, 0, stream>>>(Wq,  wqt,  512, 1024);
    k_transpose_bf16<<<dim3(2048, 1, 2), 256, 0, stream>>>(Wkl, wklt, 512, 1024);
    k_transpose_bf16<<<dim3(2048, 1, 2), 256, 0, stream>>>(Wvl, wvlt, 512, 1024);
    k_transpose_bf16<<<dim3(2048, 1, 1), 256, 0, stream>>>(Wfc, wfct, 1024, 512);

    k_to_bf16<<<dim3(4096, 1, 1), 256, 0, stream>>>(low, lowb, (long)MSd * Bd * Ld * Dd);

    k_layernorm<<<dim3(Bd * Td, 1, 1), 256, 0, stream>>>(trg, gamma, beta, xln);

    k_proj<<<dim3(8, 64, 1), 256, 0, stream>>>(xln, 0L, wqt, 0L, qb, 0);
    k_proj<<<dim3(8, 128, 2), 256, 0, stream>>>(lowb, (long)Sd * Bd * Ld * Dd,
                                                wklt, (long)1024 * 512, lkb, 1);
    k_proj<<<dim3(8, 128, 2), 256, 0, stream>>>(lowb, (long)Sd * Bd * Ld * Dd,
                                                wvlt, (long)1024 * 512, lvt, 2);

    k_gk<<<dim3(Bd * Gd, 1, 1), 256, 0, stream>>>(guider, Wkg, gk);
    k_gatt<<<dim3(128, 1, 1), 256, 0, stream>>>(qb, gk, gmask, gatt);

    k_scores<<<dim3(Td / 64, Hd, MSd * Bd), 256, 0, stream>>>(qb, lkb, gatt, lmask, fatt);
    k_av<<<dim3(Td / 64, Hd, Bd), 256, 0, stream>>>(fatt, lvt, attn);
    k_wfc<<<dim3(Dd / 128, (Bd * Td) / 64, 1), 256, 0, stream>>>(attn, wfct, trg, out);

    (void)in_sizes; (void)n_in; (void)out_size; (void)ws_size;
}
